// ASSANetEncoder_6193342841555
// MI455X (gfx1250) — matrix operand order repaired and checked by reading, not compiled, not run on hardware
//
#include <hip/hip_runtime.h>
#include <hip/hip_bf16.h>

typedef __attribute__((ext_vector_type(16))) _Float16 v16h;
typedef __attribute__((ext_vector_type(8)))  float    v8f;
typedef __attribute__((ext_vector_type(4)))  int      v4i;

#define B_   8
#define N_   16384
#define P_   4096
#define S_   32
#define CIN  64
#define CMID 128
#define COUT 128
#define EPS_ 1e-5f

// workspace byte layout
#define WS_A1 0          // 16 tiles * 1024 B  (W1*scale1, f16, WMMA A layout)
#define WS_A2 16384      // 32 tiles * 1024 B  (W2*scale2, f16, WMMA A layout)
#define WS_H1 49152      // 128 f32 biases (b1 - m1*s1)
#define WS_H2 49664      // 128 f32 biases (b2 - m2*s2)

// ---------------------------------------------------------------------------
// Pre-kernel: fold BN scale into weights, convert to f16, and pre-swizzle into
// a 16x32 A-operand layout (contiguous k, same as B) so the main kernel loads each
// A tile as plain contiguous 32 B per lane.
//   A layout used here (not AMD's table k order): lane L holds row M = L%16; half j holds K = j + 16*(L/16)
// ---------------------------------------------------------------------------
__global__ void assa_pack(const float* __restrict__ W1, const float* __restrict__ g1,
                          const float* __restrict__ b1, const float* __restrict__ m1,
                          const float* __restrict__ v1,
                          const float* __restrict__ W2, const float* __restrict__ g2,
                          const float* __restrict__ b2, const float* __restrict__ m2,
                          const float* __restrict__ v2,
                          unsigned char* __restrict__ ws) {
  int t = blockIdx.x * blockDim.x + threadIdx.x;
  _Float16* a1 = (_Float16*)(ws + WS_A1);
  _Float16* a2 = (_Float16*)(ws + WS_A2);
  float* h1 = (float*)(ws + WS_H1);
  float* h2 = (float*)(ws + WS_H2);

  if (t < 16 * 512) {                       // W1': 8 M-tiles x 2 K-steps
    int tile = t >> 9, r = t & 511;
    int L = r >> 4, j = r & 15;
    int mt = tile >> 1, kt = tile & 1;
    int o = mt * 16 + (L & 15);
    int k = kt * 32 + j + 16 * (L >> 4);
    float s = g1[o] * rsqrtf(v1[o] + EPS_);
    a1[tile * 512 + L * 16 + j] = (_Float16)(W1[o * CIN + k] * s);
  } else if (t < 16 * 512 + 32 * 512) {     // W2': 8 M-tiles x 4 K-steps
    int tt = t - 16 * 512;
    int tile = tt >> 9, r = tt & 511;
    int L = r >> 4, j = r & 15;
    int mt = tile >> 2, kt = tile & 3;
    int o = mt * 16 + (L & 15);
    int k = kt * 32 + j + 16 * (L >> 4);
    float s = g2[o] * rsqrtf(v2[o] + EPS_);
    a2[tile * 512 + L * 16 + j] = (_Float16)(W2[o * CMID + k] * s);
  } else if (t < 16 * 512 + 32 * 512 + 128) {
    int i = t - (16 * 512 + 32 * 512);
    float s = g1[i] * rsqrtf(v1[i] + EPS_);
    h1[i] = b1[i] - m1[i] * s;
  } else if (t < 16 * 512 + 32 * 512 + 256) {
    int i = t - (16 * 512 + 32 * 512 + 128);
    float s = g2[i] * rsqrtf(v2[i] + EPS_);
    h2[i] = b2[i] - m2[i] * s;
  }
}

// ---------------------------------------------------------------------------
// Main kernel: one wave32 per query point. 32 samples == WMMA N dim (2 tiles).
// GEMM1: Y1 = relu(W1'*AGG + h1)  (32 WMMAs)   -> LDS [s][k] f16
// GEMM2: Y2 = relu(W2'*Y1  + h2)  (64 WMMAs)   -> rowwise max over 32 samples
// ---------------------------------------------------------------------------
__global__ __launch_bounds__(128) void assa_main(
    const float* __restrict__ qxyz, const float* __restrict__ sxyz,
    const float* __restrict__ feat, const int* __restrict__ nidx,
    const unsigned char* __restrict__ ws, float* __restrict__ out) {

  // per-wave Y1 buffer: 32 cols x (128 + 8 pad) halves, 16-B aligned rows
  __shared__ __align__(16) _Float16 ldsY[4][S_ * 136];

  const int wid  = threadIdx.x >> 5;
  const int lane = threadIdx.x & 31;
  const int gw   = blockIdx.x * 4 + wid;   // global wave id == point id
  const int b    = gw >> 12;               // / 4096
  const int p    = gw & (P_ - 1);

  const int half = lane >> 4;              // 0: lanes 0-15, 1: lanes 16-31
  const int n    = lane & 15;              // column within 16-wide N tile

  const _Float16* a1t = (const _Float16*)(ws + WS_A1);
  const _Float16* a2t = (const _Float16*)(ws + WS_A2);
  const float*    h1  = (const float*)(ws + WS_H1);
  const float*    h2  = (const float*)(ws + WS_H2);

  const float qx = qxyz[(b * P_ + p) * 3 + 0];
  const float qy = qxyz[(b * P_ + p) * 3 + 1];
  const float qz = qxyz[(b * P_ + p) * 3 + 2];

  // the two samples this lane owns as B-matrix columns (s = n and 16+n)
  int   idxs[2];
  float rx[2], ry[2], rz[2];
#pragma unroll
  for (int h = 0; h < 2; ++h) {
    int s  = h * 16 + n;
    int id = nidx[(b * P_ + p) * S_ + s];
    idxs[h] = id;
    rx[h] = sxyz[(b * N_ + id) * 3 + 0] - qx;
    ry[h] = sxyz[(b * N_ + id) * 3 + 1] - qy;
    rz[h] = sxyz[(b * N_ + id) * 3 + 2] - qz;
  }

  // ---- build GEMM1 B operands (32x16, K=channels, N=samples) by direct gather
  // B layout: lane holds col n, halves j -> K = kt*32 + 16*half + j
  v16h B1[2][2];  // [kt][nt]
#pragma unroll
  for (int kt = 0; kt < 2; ++kt) {
#pragma unroll
    for (int nt = 0; nt < 2; ++nt) {
      const int   id = idxs[nt];
      const float ex = rx[nt], ey = ry[nt], ez = rz[nt];
      const int   c0 = kt * 32 + 16 * half;
      v16h v;
#pragma unroll
      for (int j = 0; j < 16; ++j) {
        int   c   = c0 + j;
        float pos = (c < 21) ? ex : (c < 42) ? ey : ez;  // repeat_interleave(21,21,22)
        float f   = feat[(b * CIN + c) * N_ + id];
        v[j] = (_Float16)(pos * f);
      }
      B1[kt][nt] = v;
    }
  }

  // ---- GEMM1 ----
  for (int mt = 0; mt < 8; ++mt) {
    v16h A0 = *(const v16h*)(a1t + (mt * 2 + 0) * 512 + lane * 16);
    v16h A1 = *(const v16h*)(a1t + (mt * 2 + 1) * 512 + lane * 16);
    v8f acc0, acc1;
#pragma unroll
    for (int v = 0; v < 8; ++v) {            // C/D: vgpr v -> row mt*16 + 8*half + v
      float bias = h1[mt * 16 + 8 * half + v];
      acc0[v] = bias;
      acc1[v] = bias;
    }
    acc0 = __builtin_amdgcn_wmma_f32_16x16x32_f16(false, A0, false, B1[0][0], (short)0, acc0, false, false);
    acc0 = __builtin_amdgcn_wmma_f32_16x16x32_f16(false, A1, false, B1[1][0], (short)0, acc0, false, false);
    acc1 = __builtin_amdgcn_wmma_f32_16x16x32_f16(false, A0, false, B1[0][1], (short)0, acc1, false, false);
    acc1 = __builtin_amdgcn_wmma_f32_16x16x32_f16(false, A1, false, B1[1][1], (short)0, acc1, false, false);

#pragma unroll
    for (int nt = 0; nt < 2; ++nt) {         // relu + f16, store Y1[s][k] (transposed)
      v8f a = nt ? acc1 : acc0;
      union { _Float16 hh[8]; v4i q; } u;
#pragma unroll
      for (int v = 0; v < 8; ++v) u.hh[v] = (_Float16)fmaxf(a[v], 0.0f);
      int col = nt * 16 + n;                 // sample index
      int off = col * 136 + mt * 16 + 8 * half;
      *(v4i*)&ldsY[wid][off] = u.q;          // ds_store_b128 (same-wave buffer)
    }
  }

  // ---- load GEMM2 B operands (128x16 K-major) from LDS: 16 consecutive K per lane
  v16h B2[4][2];
#pragma unroll
  for (int kt = 0; kt < 4; ++kt) {
#pragma unroll
    for (int nt = 0; nt < 2; ++nt) {
      int off = (nt * 16 + n) * 136 + kt * 32 + 16 * half;
      union { v4i q[2]; v16h h; } u;
      u.q[0] = *(const v4i*)&ldsY[wid][off];
      u.q[1] = *(const v4i*)&ldsY[wid][off + 8];
      B2[kt][nt] = u.h;
    }
  }

  // ---- GEMM2 + relu + max over samples ----
  for (int mt = 0; mt < 8; ++mt) {
    v8f acc0, acc1;
#pragma unroll
    for (int v = 0; v < 8; ++v) {
      float bias = h2[mt * 16 + 8 * half + v];
      acc0[v] = bias;
      acc1[v] = bias;
    }
#pragma unroll
    for (int kt = 0; kt < 4; ++kt) {
      v16h A = *(const v16h*)(a2t + (mt * 4 + kt) * 512 + lane * 16);
      acc0 = __builtin_amdgcn_wmma_f32_16x16x32_f16(false, A, false, B2[kt][0], (short)0, acc0, false, false);
      acc1 = __builtin_amdgcn_wmma_f32_16x16x32_f16(false, A, false, B2[kt][1], (short)0, acc1, false, false);
    }
#pragma unroll
    for (int v = 0; v < 8; ++v) {
      float m = fmaxf(fmaxf(acc0[v], 0.0f), fmaxf(acc1[v], 0.0f));
      // butterfly max across the 16 lanes of each half (wave32, warpSize==32)
      m = fmaxf(m, __shfl_xor(m, 1, 32));
      m = fmaxf(m, __shfl_xor(m, 2, 32));
      m = fmaxf(m, __shfl_xor(m, 4, 32));
      m = fmaxf(m, __shfl_xor(m, 8, 32));
      if (n == 0) {
        int o = mt * 16 + 8 * half + v;
        out[(b * COUT + o) * P_ + p] = m;
      }
    }
  }
}

extern "C" void kernel_launch(void* const* d_in, const int* in_sizes, int n_in,
                              void* d_out, int out_size, void* d_ws, size_t ws_size,
                              hipStream_t stream) {
  const float* qxyz = (const float*)d_in[0];
  const float* sxyz = (const float*)d_in[1];
  const float* feat = (const float*)d_in[2];
  const int*   nidx = (const int*)d_in[3];
  const float* W1 = (const float*)d_in[4];
  const float* g1 = (const float*)d_in[5];
  const float* b1 = (const float*)d_in[6];
  const float* m1 = (const float*)d_in[7];
  const float* v1 = (const float*)d_in[8];
  const float* W2 = (const float*)d_in[9];
  const float* g2 = (const float*)d_in[10];
  const float* b2 = (const float*)d_in[11];
  const float* m2 = (const float*)d_in[12];
  const float* v2 = (const float*)d_in[13];

  // pack weights/biases into WMMA-friendly layout in workspace
  int pack_threads = 16 * 512 + 32 * 512 + 256;
  assa_pack<<<(pack_threads + 255) / 256, 256, 0, stream>>>(
      W1, g1, b1, m1, v1, W2, g2, b2, m2, v2, (unsigned char*)d_ws);

  // one wave per point; 4 waves per 128-thread block
  assa_main<<<(B_ * P_) / 4, 128, 0, stream>>>(
      qxyz, sxyz, feat, nidx, (const unsigned char*)d_ws, (float*)d_out);
}
